// EvolveGCN_2_layer_3719441678529
// MI455X (gfx1250) — matrix operand order repaired and checked by reading, not compiled, not run on hardware
//
#include <hip/hip_runtime.h>
#include <hip/hip_bf16.h>

// ---------------------------------------------------------------------------
// EvolveGCN 2-layer for MI455X (gfx1250, wave32, WMMA f32_16x16x32_f16)
//
// T=20, N=50000, F=128, NNZ=1e6, E=5e5
// Dense matmuls via v_wmma_f32_16x16x32_f16 (f16 in, f32 acc), weights kept
// as transposed-f16 so B fragments are contiguous 32B loads per lane.
// SpMM via L2-resident f32 atomics. Top-k via 200KB dynamic LDS (320KB/WGP).
// ---------------------------------------------------------------------------

#define TT 20
#define NN 50000
#define FF 128
#define NNZ 1000000
#define EE 500000

typedef __attribute__((ext_vector_type(16))) _Float16 v16h;
typedef __attribute__((ext_vector_type(8)))  float    v8f;

// ---- WMMA fragment helpers (wave32; contiguous K order, used by A and B alike) ----------
// A (16x32 f16): lane = 16h+m holds row m; K = {16h..16h+15}
__device__ __forceinline__ v16h load_a_row_f32(const float* __restrict__ prow,
                                               int k0, int h) {
  v16h a;
#pragma unroll
  for (int e = 0; e < 8; ++e) a[e]     = (_Float16)prow[k0 + 16*h + e];
#pragma unroll
  for (int e = 0; e < 8; ++e) a[8 + e] = (_Float16)prow[k0 + 16*h + 8 + e];
  return a;
}

__device__ __forceinline__ v8f wmma_f16(v16h a, v16h b, v8f c) {
  return __builtin_amdgcn_wmma_f32_16x16x32_f16(false, a, false, b,
                                                (short)0, c, false, false);
}

// ---------------------------------------------------------------------------
// ||p|| for both layers
__global__ void pnorm_kernel(const float* __restrict__ p1,
                             const float* __restrict__ p2,
                             float* __restrict__ pn) {
  __shared__ float sm[128];
  int t = threadIdx.x;
  sm[t] = p1[t] * p1[t];
  __syncthreads();
  for (int s = 64; s > 0; s >>= 1) { if (t < s) sm[t] += sm[t + s]; __syncthreads(); }
  if (t == 0) pn[0] = sqrtf(sm[0]);
  __syncthreads();
  sm[t] = p2[t] * p2[t];
  __syncthreads();
  for (int s = 64; s > 0; s >>= 1) { if (t < s) sm[t] += sm[t + s]; __syncthreads(); }
  if (t == 0) pn[1] = sqrtf(sm[0]);
}

// W_init -> W (f32) and WT16 (transposed f16 for B operand)
__global__ void prepw_kernel(const float* __restrict__ Wi,
                             float* __restrict__ W, _Float16* __restrict__ WT16) {
  int i = blockIdx.x * 256 + threadIdx.x;          // 16384 threads
  int r = i >> 7, c = i & 127;
  float v = Wi[i];
  W[i] = v;
  WT16[c * 128 + r] = (_Float16)v;
}

__global__ void zero_kernel(float* __restrict__ p, int n) {
  int i = blockIdx.x * 256 + threadIdx.x;
  if (i < n) p[i] = 0.0f;
}

// y[i] = (H[i,:] . p) / ||p||  — one wave32 per node
__global__ void ydot_kernel(const float* __restrict__ H, const float* __restrict__ p,
                            const float* __restrict__ pn, float* __restrict__ y, int n) {
  int wave = (int)((blockIdx.x * (unsigned)blockDim.x + threadIdx.x) >> 5);
  int lane = threadIdx.x & 31;
  if (wave >= n) return;
  float4 h4 = *(const float4*)(H + (size_t)wave * FF + lane * 4);
  float4 p4 = *(const float4*)(p + lane * 4);
  float s = h4.x * p4.x + h4.y * p4.y + h4.z * p4.z + h4.w * p4.w;
#pragma unroll
  for (int o = 16; o > 0; o >>= 1) s += __shfl_down(s, o, 32);
  if (lane == 0) y[wave] = s / pn[0];
}

// Deterministic exact top-128 with smallest-index tie-break; builds
// XtB[j][f] = H[idx_j][f] * yv_j  (f16, row-major = B operand layout).
// y kept in dynamic LDS (200KB fits the 320KB WGP LDS).
__global__ __launch_bounds__(1024) void topk_xt_kernel(
    const float* __restrict__ y_in, const float* __restrict__ H,
    _Float16* __restrict__ XtB, int n) {
  extern __shared__ float smem[];
  float* ys = smem;
  int npad = ((n + 511) / 512) * 512;
  unsigned long long* red = (unsigned long long*)(smem + npad);
  int tid = threadIdx.x;
  for (int i = tid; i < n; i += 1024) ys[i] = y_in[i];
  __syncthreads();
  for (int j = 0; j < FF; ++j) {
    unsigned long long best = 0ull;
    for (int i = tid; i < n; i += 1024) {
      unsigned int b = __float_as_uint(ys[i]);
      unsigned int k = b ^ ((b & 0x80000000u) ? 0xFFFFFFFFu : 0x80000000u);
      unsigned long long key =
          ((unsigned long long)k << 32) | (unsigned long long)(0xFFFFFFFFu - (unsigned)i);
      best = key > best ? key : best;
    }
    red[tid] = best;
    __syncthreads();
    for (int s = 512; s > 0; s >>= 1) {
      if (tid < s && red[tid + s] > red[tid]) red[tid] = red[tid + s];
      __syncthreads();
    }
    unsigned long long win = red[0];
    int idx = (int)(0xFFFFFFFFu - (unsigned)(win & 0xFFFFFFFFull));
    unsigned int kb = (unsigned)(win >> 32);
    unsigned int ob = (kb & 0x80000000u) ? (kb ^ 0x80000000u) : (kb ^ 0xFFFFFFFFu);
    float yv = __uint_as_float(ob);
    if (tid < FF) XtB[j * FF + tid] = (_Float16)(H[(size_t)idx * FF + tid] * yv);
    if (tid == 0) ys[idx] = -__builtin_inff();
    __syncthreads();
  }
}

// ---------------------------------------------------------------------------
// GRU stage 1: Z = sigmoid(Wz@Xt + Uz@W + Bz) (f32 to ws),
//              RWT16 = transpose(sigmoid(Wr@Xt+Ur@W+Br) * W) as f16.
// One workgroup, 8 waves, each wave owns 16 output rows; WMMA everywhere.
__global__ __launch_bounds__(256) void gru_zr_kernel(
    const float* __restrict__ Wz, const float* __restrict__ Uz, const float* __restrict__ Bz,
    const float* __restrict__ Wr, const float* __restrict__ Ur, const float* __restrict__ Br,
    const _Float16* __restrict__ XtB, const _Float16* __restrict__ WoldT16,
    const float* __restrict__ Wold, float* __restrict__ Zs, _Float16* __restrict__ RWT16) {
  int tid = threadIdx.x, wave = tid >> 5, lane = tid & 31;
  int h = lane >> 4, mn = lane & 15;
  int rowb = wave * 16;
  const float* wzr = Wz + (size_t)(rowb + mn) * FF;
  const float* uzr = Uz + (size_t)(rowb + mn) * FF;
  const float* wrr = Wr + (size_t)(rowb + mn) * FF;
  const float* urr = Ur + (size_t)(rowb + mn) * FF;
  for (int c = 0; c < 8; ++c) {
    int cb = c * 16;
    v8f az, ar;
#pragma unroll
    for (int r = 0; r < 8; ++r) {
      int orow = rowb + r + 8 * h, col = cb + mn;
      az[r] = Bz[orow * FF + col];
      ar[r] = Br[orow * FF + col];
    }
#pragma unroll
    for (int kc = 0; kc < 4; ++kc) {
      int k0 = kc * 32;
      v16h bx = *(const v16h*)(&XtB[(size_t)(cb + mn) * FF + k0 + 16 * h]);
      v16h bw = *(const v16h*)(&WoldT16[(size_t)(cb + mn) * FF + k0 + 16 * h]);
      az = wmma_f16(load_a_row_f32(wzr, k0, h), bx, az);
      az = wmma_f16(load_a_row_f32(uzr, k0, h), bw, az);
      ar = wmma_f16(load_a_row_f32(wrr, k0, h), bx, ar);
      ar = wmma_f16(load_a_row_f32(urr, k0, h), bw, ar);
    }
#pragma unroll
    for (int r = 0; r < 8; ++r) {
      int orow = rowb + r + 8 * h, col = cb + mn;
      float z  = 1.0f / (1.0f + expf(-az[r]));
      float rg = 1.0f / (1.0f + expf(-ar[r]));
      Zs[orow * FF + col] = z;
      RWT16[col * FF + orow] = (_Float16)(rg * Wold[orow * FF + col]);
    }
  }
}

// GRU stage 2: Ht = tanh(Wh@Xt + Uh@(R*W) + Bh); W = (1-Z)*W + Z*Ht.
// Writes updated W (f32) and WT16 (transposed f16).
__global__ __launch_bounds__(256) void gru_h_kernel(
    const float* __restrict__ Wh, const float* __restrict__ Uh, const float* __restrict__ Bh,
    const _Float16* __restrict__ XtB, const _Float16* __restrict__ RWT16,
    const float* __restrict__ Zs, float* __restrict__ W, _Float16* __restrict__ WT16) {
  int tid = threadIdx.x, wave = tid >> 5, lane = tid & 31;
  int h = lane >> 4, mn = lane & 15;
  int rowb = wave * 16;
  const float* whr = Wh + (size_t)(rowb + mn) * FF;
  const float* uhr = Uh + (size_t)(rowb + mn) * FF;
  for (int c = 0; c < 8; ++c) {
    int cb = c * 16;
    v8f ah;
#pragma unroll
    for (int r = 0; r < 8; ++r) ah[r] = Bh[(rowb + r + 8 * h) * FF + cb + mn];
#pragma unroll
    for (int kc = 0; kc < 4; ++kc) {
      int k0 = kc * 32;
      v16h bx = *(const v16h*)(&XtB[(size_t)(cb + mn) * FF + k0 + 16 * h]);
      v16h bw = *(const v16h*)(&RWT16[(size_t)(cb + mn) * FF + k0 + 16 * h]);
      ah = wmma_f16(load_a_row_f32(whr, k0, h), bx, ah);
      ah = wmma_f16(load_a_row_f32(uhr, k0, h), bw, ah);
    }
#pragma unroll
    for (int r = 0; r < 8; ++r) {
      int orow = rowb + r + 8 * h, col = cb + mn;
      float ht = tanhf(ah[r]);
      float z  = Zs[orow * FF + col];
      float wn = (1.0f - z) * W[orow * FF + col] + z * ht;
      W[orow * FF + col] = wn;
      WT16[col * FF + orow] = (_Float16)wn;
    }
  }
}

// ---------------------------------------------------------------------------
// SpMM scatter: S[rows[i],:] += vals[i] * H[cols[i],:]  (f32 atomics, L2-hot)
__global__ void spmm_kernel(const int* __restrict__ rows, const int* __restrict__ cols,
                            const float* __restrict__ vals, const float* __restrict__ H,
                            float* __restrict__ S, int nnz) {
  long long gid = (long long)blockIdx.x * blockDim.x + threadIdx.x;
  if (gid >= (long long)nnz * 32) return;
  int i  = (int)(gid >> 5);
  int fc = (int)(gid & 31) * 4;
  float v = vals[i];
  int r = rows[i], c = cols[i];
  float4 h4 = *(const float4*)(H + (size_t)c * FF + fc);
  float* sp = S + (size_t)r * FF + fc;
  atomicAdd(sp + 0, v * h4.x);
  atomicAdd(sp + 1, v * h4.y);
  atomicAdd(sp + 2, v * h4.z);
  atomicAdd(sp + 3, v * h4.w);
}

// ---------------------------------------------------------------------------
// D[M,128] = act(S[M,128] @ W) with W given as WT16[j][k]=W[k][j] (f16).
// Block = 256 thr = 8 waves; each wave does 16 rows x 128 cols (8 C tiles).
// WT16 staged in 32KB LDS; A fragments converted f32->f16 on load.
__global__ __launch_bounds__(256) void mm_wmma_kernel(
    const float* __restrict__ S, const _Float16* __restrict__ WT16,
    float* __restrict__ D, int M, int relu) {
  __shared__ _Float16 Bs[FF * FF];
  int tid = threadIdx.x;
  for (int i = tid * 8; i < FF * FF; i += 256 * 8)
    *(float4*)(&Bs[i]) = *(const float4*)(&WT16[i]);
  __syncthreads();

  int wave = tid >> 5, lane = tid & 31;
  int h = lane >> 4, mn = lane & 15;
  int rowb = blockIdx.x * 128 + wave * 16;
  int arow = rowb + mn;                 // A-side row for this lane (both halves)
  bool valid = arow < M;
  const float* pr = S + (size_t)(valid ? arow : 0) * FF;

  v8f acc[8];
#pragma unroll
  for (int c = 0; c < 8; ++c)
#pragma unroll
    for (int r = 0; r < 8; ++r) acc[c][r] = 0.0f;

#pragma unroll
  for (int kc = 0; kc < 4; ++kc) {
    int k0 = kc * 32;
    v16h a;
#pragma unroll
    for (int e = 0; e < 8; ++e)
      a[e] = valid ? (_Float16)pr[k0 + 16 * h + e] : (_Float16)0.0f;
#pragma unroll
    for (int e = 0; e < 8; ++e)
      a[8 + e] = valid ? (_Float16)pr[k0 + 16 * h + 8 + e] : (_Float16)0.0f;
#pragma unroll
    for (int c = 0; c < 8; ++c) {
      v16h b = *(const v16h*)(&Bs[(c * 16 + mn) * FF + k0 + 16 * h]);
      acc[c] = wmma_f16(a, b, acc[c]);
    }
  }
#pragma unroll
  for (int c = 0; c < 8; ++c) {
    int col = c * 16 + mn;
#pragma unroll
    for (int r = 0; r < 8; ++r) {
      int orow = rowb + r + 8 * h;
      if (orow < M) {
        float v = acc[c][r];
        if (relu) v = v > 0.0f ? v : 0.0f;
        D[(size_t)orow * FF + col] = v;
      }
    }
  }
}

// ---------------------------------------------------------------------------
// out[e,:] = concat(Yf[src_e], Yf[trg_e]) @ U   (U: [256,2], staged in LDS)
__global__ void edge_kernel(const float* __restrict__ Y, const int* __restrict__ edges,
                            const float* __restrict__ U, float* __restrict__ out) {
  __shared__ float Us[512];
  for (int i = threadIdx.x; i < 512; i += 256) Us[i] = U[i];
  __syncthreads();
  int e = blockIdx.x * 256 + threadIdx.x;
  if (e >= EE) return;
  int t = edges[e], s = edges[EE + e], d = edges[2 * EE + e];
  const float* ys = Y + ((size_t)t * NN + s) * FF;
  const float* yt = Y + ((size_t)t * NN + d) * FF;
  float a0 = 0.0f, a1 = 0.0f;
#pragma unroll 4
  for (int k = 0; k < FF; k += 4) {
    float4 v = *(const float4*)(ys + k);
    a0 += v.x * Us[(k + 0) * 2] + v.y * Us[(k + 1) * 2] +
          v.z * Us[(k + 2) * 2] + v.w * Us[(k + 3) * 2];
    a1 += v.x * Us[(k + 0) * 2 + 1] + v.y * Us[(k + 1) * 2 + 1] +
          v.z * Us[(k + 2) * 2 + 1] + v.w * Us[(k + 3) * 2 + 1];
  }
#pragma unroll 4
  for (int k = 0; k < FF; k += 4) {
    float4 v = *(const float4*)(yt + k);
    int kk = 128 + k;
    a0 += v.x * Us[(kk + 0) * 2] + v.y * Us[(kk + 1) * 2] +
          v.z * Us[(kk + 2) * 2] + v.w * Us[(kk + 3) * 2];
    a1 += v.x * Us[(kk + 0) * 2 + 1] + v.y * Us[(kk + 1) * 2 + 1] +
          v.z * Us[(kk + 2) * 2 + 1] + v.w * Us[(kk + 3) * 2 + 1];
  }
  out[(size_t)e * 2 + 0] = a0;
  out[(size_t)e * 2 + 1] = a1;
}

__global__ void copyw_kernel(const float* __restrict__ W1, const float* __restrict__ W2,
                             float* __restrict__ dst) {
  int i = blockIdx.x * 256 + threadIdx.x;   // 32768
  if (i < 16384) dst[i] = W1[i];
  else           dst[i] = W2[i - 16384];
}

// ---------------------------------------------------------------------------
extern "C" void kernel_launch(void* const* d_in, const int* in_sizes, int n_in,
                              void* d_out, int out_size, void* d_ws, size_t ws_size,
                              hipStream_t stream) {
  (void)in_sizes; (void)n_in; (void)out_size; (void)ws_size;
  const float* X      = (const float*)d_in[0];
  const int*   A_rows = (const int*)  d_in[1];
  const int*   A_cols = (const int*)  d_in[2];
  const float* A_vals = (const float*)d_in[3];
  const int*   edges  = (const int*)  d_in[4];
  const float* p1     = (const float*)d_in[5];
  const float* W_Z  = (const float*)d_in[6],  *U_Z  = (const float*)d_in[7],  *B_Z  = (const float*)d_in[8];
  const float* W_R  = (const float*)d_in[9],  *U_R  = (const float*)d_in[10], *B_R  = (const float*)d_in[11];
  const float* W_H  = (const float*)d_in[12], *U_H  = (const float*)d_in[13], *B_H  = (const float*)d_in[14];
  const float* Winit  = (const float*)d_in[15];
  const float* p2     = (const float*)d_in[16];
  const float* W_Z2 = (const float*)d_in[17], *U_Z2 = (const float*)d_in[18], *B_Z2 = (const float*)d_in[19];
  const float* W_R2 = (const float*)d_in[20], *U_R2 = (const float*)d_in[21], *B_R2 = (const float*)d_in[22];
  const float* W_H2 = (const float*)d_in[23], *U_H2 = (const float*)d_in[24], *B_H2 = (const float*)d_in[25];
  const float* Winit2 = (const float*)d_in[26];
  const float* U      = (const float*)d_in[27];
  float* out = (float*)d_out;

  // ---- workspace carve-up (256B aligned) ----
  char* w = (char*)d_ws;
  auto carve = [&](size_t bytes) -> char* {
    char* pp = w;
    w += (bytes + 255) & ~(size_t)255;
    return pp;
  };
  float*    Y     = (float*)   carve((size_t)TT * NN * FF * 4);  // 512 MB
  float*    Sbuf  = (float*)   carve((size_t)NN * FF * 4);       // spmm accum
  float*    Hbuf  = (float*)   carve((size_t)NN * FF * 4);       // layer-1 output
  float*    ybuf  = (float*)   carve((size_t)NN * 4);
  float*    W1    = (float*)   carve(16384 * 4);
  float*    W2    = (float*)   carve(16384 * 4);
  float*    Zs    = (float*)   carve(16384 * 4);
  _Float16* W1T16 = (_Float16*)carve(16384 * 2);
  _Float16* W2T16 = (_Float16*)carve(16384 * 2);
  _Float16* XtB16 = (_Float16*)carve(16384 * 2);
  _Float16* RWT16 = (_Float16*)carve(16384 * 2);
  float*    pn    = (float*)   carve(256);

  const int NF = NN * FF;
  const dim3 b256(256);
  const int gZero = (NF + 255) / 256;
  const int gY    = (NN * 32 + 255) / 256;       // one wave per node
  const int gSp   = (int)(((long long)NNZ * 32 + 255) / 256);
  const int gMM   = (NN + 127) / 128;
  const size_t topkShm = ((size_t)((NN + 511) / 512) * 512) * 4 + 1024 * 8;

  pnorm_kernel<<<1, 128, 0, stream>>>(p1, p2, pn);
  prepw_kernel<<<64, b256, 0, stream>>>(Winit,  W1, W1T16);
  prepw_kernel<<<64, b256, 0, stream>>>(Winit2, W2, W2T16);

  for (int t = 0; t < TT; ++t) {
    const float* Xt   = X      + (size_t)t * NF;
    const int*   rows = A_rows + (size_t)t * NNZ;
    const int*   cols = A_cols + (size_t)t * NNZ;
    const float* vals = A_vals + (size_t)t * NNZ;

    // --- layer 1: W1 = GRU(Xt, W1) ---
    ydot_kernel<<<gY, b256, 0, stream>>>(Xt, p1, pn + 0, ybuf, NN);
    topk_xt_kernel<<<1, 1024, topkShm, stream>>>(ybuf, Xt, XtB16, NN);
    gru_zr_kernel<<<1, b256, 0, stream>>>(W_Z, U_Z, B_Z, W_R, U_R, B_R,
                                          XtB16, W1T16, W1, Zs, RWT16);
    gru_h_kernel<<<1, b256, 0, stream>>>(W_H, U_H, B_H, XtB16, RWT16, Zs, W1, W1T16);

    // --- H = relu(spmm(A, Xt) @ W1_new) ---
    zero_kernel<<<gZero, b256, 0, stream>>>(Sbuf, NF);
    spmm_kernel<<<gSp, b256, 0, stream>>>(rows, cols, vals, Xt, Sbuf, NNZ);
    mm_wmma_kernel<<<gMM, b256, 0, stream>>>(Sbuf, W1T16, Hbuf, NN, /*relu=*/1);

    // --- layer 2: W2 = GRU(H, W2) ---
    ydot_kernel<<<gY, b256, 0, stream>>>(Hbuf, p2, pn + 1, ybuf, NN);
    topk_xt_kernel<<<1, 1024, topkShm, stream>>>(ybuf, Hbuf, XtB16, NN);
    gru_zr_kernel<<<1, b256, 0, stream>>>(W_Z2, U_Z2, B_Z2, W_R2, U_R2, B_R2,
                                          XtB16, W2T16, W2, Zs, RWT16);
    gru_h_kernel<<<1, b256, 0, stream>>>(W_H2, U_H2, B_H2, XtB16, RWT16, Zs, W2, W2T16);

    // --- Y[t] = spmm(A, H) @ W2_new ---
    zero_kernel<<<gZero, b256, 0, stream>>>(Sbuf, NF);
    spmm_kernel<<<gSp, b256, 0, stream>>>(rows, cols, vals, Hbuf, Sbuf, NNZ);
    mm_wmma_kernel<<<gMM, b256, 0, stream>>>(Sbuf, W2T16, Y + (size_t)t * NF, NN, /*relu=*/0);
  }

  // --- edge readout + final weights ---
  edge_kernel<<<(EE + 255) / 256, b256, 0, stream>>>(Y, edges, U, out);
  copyw_kernel<<<128, b256, 0, stream>>>(W1, W2, out + (size_t)EE * 2);
}
